// RepoModel_55181739819296
// MI455X (gfx1250) — matrix operand order repaired and checked by reading, not compiled, not run on hardware
//
#include <hip/hip_runtime.h>

// ---------------------------------------------------------------------------
// Model constants (match reference)
// ---------------------------------------------------------------------------
#define EMB   128
#define LDESC 32
#define NTAG  8
#define NTOK  3
#define F1F2  96
#define DCAT  192
#define DMID  192
#define DOUT  128

typedef __attribute__((ext_vector_type(16))) _Float16 v16h;
typedef __attribute__((ext_vector_type(8)))  _Float16 v8h;
typedef __attribute__((ext_vector_type(4)))  _Float16 v4h;
typedef __attribute__((ext_vector_type(8)))  float    v8f;

__device__ __forceinline__ v8f wmma_f16(v16h a, v16h b, v8f c) {
  // D = A(16x32 f16) * B(32x16 f16) + C(16x16 f32)
  return __builtin_amdgcn_wmma_f32_16x16x32_f16(
      /*neg_a=*/false, a, /*neg_b=*/false, b,
      /*c_mod=*/(short)0, c, /*reuse_a=*/false, /*reuse_b=*/false);
}

// A-fragment (16x32 f16) per ISA layout:
// lanes 0-15: M=lane,    K = k0+[0..7] and k0+16+[0..7]
// lanes 16-31: M=lane-16, K = k0+8+[0..7] and k0+24+[0..7]
// For conv windows A[p][k] = xe[p*128 + k] (k = kh*128+e collapses).
__device__ __forceinline__ v16h lds_a_frag(const _Float16* mat, int m0, int k0, int lane) {
  const _Float16* p = mat + (m0 + (lane & 15)) * 128 + k0 + ((lane >> 4) << 3);
  v8h a0 = *(const v8h*)(p);
  v8h a1 = *(const v8h*)(p + 16);
  v16h r;
#pragma unroll
  for (int i = 0; i < 8; ++i) { r[i] = a0[i]; r[i + 8] = a1[i]; }
  return r;
}

// A-fragment from a global row-major f16 matrix [nrows][ldk]
__device__ __forceinline__ v16h glb_a_frag(const _Float16* row /*= A + m*ldk + h8*/, int k0) {
  v8h a0 = *(const v8h*)(row + k0);
  v8h a1 = *(const v8h*)(row + k0 + 16);
  v16h r;
#pragma unroll
  for (int i = 0; i < 8; ++i) { r[i] = a0[i]; r[i + 8] = a1[i]; }
  return r;
}

// B-fragment (32x16 f16): per lane, column n = n0+(lane&15),
// K = k0+8*(lane>>4)+[0..7] and k0+16+8*(lane>>4)+[0..7]. Wt is [ncols][kdim] row-major f16.
__device__ __forceinline__ v16h glb_b_frag(const _Float16* W, int ldk, int n0, int k0, int lane) {
  const _Float16* p = W + (size_t)(n0 + (lane & 15)) * ldk + k0 + ((lane >> 4) << 3);
  v8h b0 = *(const v8h*)(p);
  v8h b1 = *(const v8h*)(p + 16);
  v16h r;
#pragma unroll
  for (int i = 0; i < 8; ++i) { r[i] = b0[i]; r[i + 8] = b1[i]; }
  return r;
}

// ---------------------------------------------------------------------------
// Weight / embedding conversion
// ---------------------------------------------------------------------------
__global__ void cvt_f16_kernel(const float* __restrict__ s, _Float16* __restrict__ d, int n) {
  int i = blockIdx.x * blockDim.x + threadIdx.x;
  if (i < n) d[i] = (_Float16)s[i];
}

__global__ void cvt_f16_t_kernel(const float* __restrict__ s, _Float16* __restrict__ d,
                                 int rows, int cols) { // s[rows][cols] -> d[cols][rows]
  int i = blockIdx.x * blockDim.x + threadIdx.x;
  if (i < rows * cols) {
    int r = i / cols, c = i - r * cols;
    d[(size_t)c * rows + r] = (_Float16)s[i];
  }
}

__global__ void relu_cvt_kernel(const float* __restrict__ a, _Float16* __restrict__ d, int n) {
  int i = blockIdx.x * blockDim.x + threadIdx.x;
  if (i < n) { float v = a[i]; d[i] = (_Float16)(v > 0.f ? v : 0.f); }
}

// ---------------------------------------------------------------------------
// Per-node fused TextCNN + tag-FC kernel.  One workgroup (256 thr = 8 waves)
// per node.  36 WMMA output tiles split across 8 waves (wave id scalarized).
// ---------------------------------------------------------------------------
__global__ void __launch_bounds__(256)
node_features_kernel(const int* __restrict__ feats, const int* __restrict__ tags,
                     const _Float16* __restrict__ embed_h,
                     const _Float16* __restrict__ w1h, const _Float16* __restrict__ w2h,
                     const _Float16* __restrict__ fcwh,
                     const float* __restrict__ b1, const float* __restrict__ b2,
                     const float* __restrict__ bn1g, const float* __restrict__ bn1b,
                     const float* __restrict__ bn1m, const float* __restrict__ bn1v,
                     const float* __restrict__ bn2g, const float* __restrict__ bn2b,
                     const float* __restrict__ bn2m, const float* __restrict__ bn2v,
                     _Float16* __restrict__ xh)
{
  __shared__ _Float16 xe[36 * 128];   // desc embeddings, rows 32..35 zero pad
  __shared__ _Float16 ts[16 * 128];   // tag sums, rows 8..15 zero pad
  __shared__ float cmax[DCAT];
  __shared__ float tsum[DCAT];

  const int n = blockIdx.x;
  const int tid = threadIdx.x;

  if (tid < DCAT) { cmax[tid] = 0.f; tsum[tid] = 0.f; }

  // xe fill: 36 rows x 32 chunks of 4 halves (8B vector moves)
  for (int i = tid; i < 36 * 32; i += 256) {
    int r = i >> 5, cc = (i & 31) << 2;
    v4h v = {};
    if (r < LDESC)
      v = *(const v4h*)(embed_h + (size_t)feats[n * LDESC + r] * EMB + cc);
    *(v4h*)(xe + r * 128 + cc) = v;
  }
  // ts fill: 16 rows x 32 chunks; rows 0-7 = sum of 3 token embeds
  for (int i = tid; i < 16 * 32; i += 256) {
    int t = i >> 5, cc = (i & 31) << 2;
    v4h s = {};
    if (t < NTAG) {
      const int* tp = tags + ((size_t)n * NTAG + t) * NTOK;
      v4h e0 = *(const v4h*)(embed_h + (size_t)tp[0] * EMB + cc);
      v4h e1 = *(const v4h*)(embed_h + (size_t)tp[1] * EMB + cc);
      v4h e2 = *(const v4h*)(embed_h + (size_t)tp[2] * EMB + cc);
#pragma unroll
      for (int j = 0; j < 4; ++j)
        s[j] = (_Float16)((float)e0[j] + (float)e1[j] + (float)e2[j]);
    }
    *(v4h*)(ts + t * 128 + cc) = s;
  }
  __syncthreads();

  const int lane = tid & 31;
  const int wave = __builtin_amdgcn_readfirstlane(tid >> 5);  // SGPR -> scalar CF
  // units 0..11: conv1 tiles, 12..23: conv2 tiles, 24..35: tag tiles
  for (int u = wave; u < 36; u += 8) {
    v8f c = {};
    if (u < 24) {
      const int cv = (u >= 12);
      const int t  = cv ? u - 12 : u;
      const int m0 = (t >= 6) ? 16 : 0;
      const int n0 = (t % 6) << 4;
      const _Float16* W = cv ? w2h : w1h;
      const int ldk   = cv ? 512 : 256;
      const int kstep = cv ? 16  : 8;
      for (int k = 0; k < kstep; ++k) {
        v16h a = lds_a_frag(xe, m0, k * 32, lane);
        v16h b = glb_b_frag(W, ldk, n0, k * 32, lane);
        c = wmma_f16(a, b, c);
      }
      const float* bias = cv ? b2 : b1;
      const int pmax = cv ? 28 : 30;        // valid conv positions
      const int foff = cv ? 96 : 0;
      float bv = bias[n0 + (lane & 15)];
      int mhi = (lane >> 4) << 3;
      float mx = 0.f;                        // relu >= 0 so 0 is safe identity
#pragma unroll
      for (int r = 0; r < 8; ++r) {
        int p = m0 + mhi + r;
        float vv = c[r] + bv;
        vv = vv > 0.f ? vv : 0.f;
        if (p <= pmax && vv > mx) mx = vv;
      }
      // float bits compare monotonically for values >= 0
      atomicMax((int*)&cmax[foff + n0 + (lane & 15)], __float_as_int(mx));
    } else {
      const int n0 = (u - 24) << 4;          // over concat(fc1,fc2) cols
      for (int k = 0; k < 4; ++k) {
        v16h a = lds_a_frag(ts, 0, k * 32, lane);
        v16h b = glb_b_frag(fcwh, 128, n0, k * 32, lane);
        c = wmma_f16(a, b, c);
      }
      float s = 0.f;
#pragma unroll
      for (int r = 0; r < 8; ++r) { float vv = c[r]; s += vv > 0.f ? vv : 0.f; }
      atomicAdd(&tsum[n0 + (lane & 15)], s);
    }
  }
  __syncthreads();

  if (tid < DCAT) {
    float dsp = cmax[tid];
    dsp = (dsp - bn1m[tid]) * (bn1g[tid] * rsqrtf(bn1v[tid] + 1e-5f)) + bn1b[tid];
    float tg = tsum[tid];
    tg = (tg - bn2m[tid]) * (bn2g[tid] * rsqrtf(bn2v[tid] + 1e-5f)) + bn2b[tid];
    xh[(size_t)n * DCAT + tid] = (_Float16)(dsp + tg);
  }
}

// ---------------------------------------------------------------------------
// GCN dense part: Y[nrows,dout] = A[nrows,din] @ W  (A f16, W pre-transposed
// f16).  One wave per 16x32 output tile: two WMMAs share each A fragment.
// ---------------------------------------------------------------------------
__global__ void __launch_bounds__(256)
gcn_gemm_kernel(const _Float16* __restrict__ Ah, const _Float16* __restrict__ Wt,
                float* __restrict__ Y, int nrows, int din, int dout)
{
  const int npair  = dout >> 5;                 // 16x32 tiles per row band
  const int mtiles = (nrows + 15) >> 4;
  const int total  = mtiles * npair;
  const int tile = blockIdx.x * (blockDim.x >> 5) +
                   __builtin_amdgcn_readfirstlane(threadIdx.x >> 5);
  if (tile >= total) return;
  const int lane = threadIdx.x & 31;
  const int m0 = (tile / npair) << 4;
  const int n0 = (tile % npair) << 5;

  int mrow = m0 + (lane & 15);
  if (mrow >= nrows) mrow = nrows - 1;          // clamp (stores guarded below)
  const _Float16* arow = Ah + (size_t)mrow * din + ((lane >> 4) << 3);

  v8f c0 = {}, c1 = {};
  for (int k0 = 0; k0 < din; k0 += 32) {
    v16h a  = glb_a_frag(arow, k0);
    v16h b0 = glb_b_frag(Wt, din, n0,      k0, lane);
    v16h b1 = glb_b_frag(Wt, din, n0 + 16, k0, lane);
    c0 = wmma_f16(a, b0, c0);
    c1 = wmma_f16(a, b1, c1);
  }
  const int mhi = (lane >> 4) << 3;
  const int col = n0 + (lane & 15);
#pragma unroll
  for (int r = 0; r < 8; ++r) {
    int m = m0 + mhi + r;
    if (m < nrows) {
      Y[(size_t)m * dout + col]      = c0[r];
      Y[(size_t)m * dout + col + 16] = c1[r];
    }
  }
}

// ---------------------------------------------------------------------------
// Graph side: degrees, self-loop init, edge scatter, L2 norm, gather
// ---------------------------------------------------------------------------
__global__ void deg_init_kernel(float* deg, int n) {
  int i = blockIdx.x * blockDim.x + threadIdx.x;
  if (i < n) deg[i] = 1.f;                    // self-loop
}
__global__ void deg_scatter_kernel(const int* __restrict__ dst, float* deg, int ne) {
  int i = blockIdx.x * blockDim.x + threadIdx.x;
  if (i < ne) atomicAdd(&deg[dst[i]], 1.f);
}
__global__ void rsqrt_kernel(float* deg, int n) {
  int i = blockIdx.x * blockDim.x + threadIdx.x;
  if (i < n) deg[i] = rsqrtf(deg[i]);
}
__global__ void agg_init_kernel(float* __restrict__ agg, const float* __restrict__ hp,
                                const float* __restrict__ dinv, const float* __restrict__ bias,
                                int nrows, int dout) {
  int i = blockIdx.x * blockDim.x + threadIdx.x;
  if (i >= nrows * dout) return;
  int n = i / dout, j = i - n * dout;
  float di = dinv[n];
  agg[i] = hp[i] * di * di + bias[j];         // self-loop message + bias
}
__global__ void edge_scatter_kernel(const float* __restrict__ hp, float* __restrict__ agg,
                                    const int* __restrict__ src, const int* __restrict__ dst,
                                    const float* __restrict__ dinv, int ne, int dout) {
  int e = blockIdx.x * (blockDim.x >> 5) +
          __builtin_amdgcn_readfirstlane(threadIdx.x >> 5);  // scalar edge id
  if (e >= ne) return;
  int lane = threadIdx.x & 31;
  int s = src[e], d = dst[e];
  float coef = dinv[s] * dinv[d];
  const float* hs = hp + (size_t)s * dout;
  float* ad = agg + (size_t)d * dout;
  for (int j = lane; j < dout; j += 32)
    atomicAdd(&ad[j], hs[j] * coef);
}
__global__ void l2norm_kernel(float* __restrict__ h, int nrows) {
  int node = blockIdx.x * (blockDim.x >> 5) +
             __builtin_amdgcn_readfirstlane(threadIdx.x >> 5);
  if (node >= nrows) return;
  int lane = threadIdx.x & 31;
  float* row = h + (size_t)node * DOUT;
  float v[4]; float ss = 0.f;
#pragma unroll
  for (int i = 0; i < 4; ++i) { v[i] = row[lane + 32 * i]; ss += v[i] * v[i]; }
#pragma unroll
  for (int m = 16; m >= 1; m >>= 1) ss += __shfl_xor(ss, m, 32);
  float s = 1.f / fmaxf(sqrtf(ss), 1e-12f);
#pragma unroll
  for (int i = 0; i < 4; ++i) row[lane + 32 * i] = v[i] * s;
}
__global__ void gather_out_kernel(const float* __restrict__ hn, const int* __restrict__ bridge_ids,
                                  const int* __restrict__ repo_index, float* __restrict__ out,
                                  int nb, int bk) {
  int total = (nb + bk) * DOUT;
  for (int i = blockIdx.x * blockDim.x + threadIdx.x; i < total;
       i += gridDim.x * blockDim.x) {
    int row = i >> 7, j = i & 127;
    int node = (row < nb) ? bridge_ids[row] : repo_index[row - nb];
    out[i] = hn[(size_t)node * DOUT + j];
  }
}

// ---------------------------------------------------------------------------
// Host launcher
// ---------------------------------------------------------------------------
extern "C" void kernel_launch(void* const* d_in, const int* in_sizes, int n_in,
                              void* d_out, int out_size, void* d_ws, size_t ws_size,
                              hipStream_t stream) {
  (void)n_in; (void)out_size; (void)ws_size;
  const int*   repo_index = (const int*)d_in[0];
  const int*   feats      = (const int*)d_in[1];
  const int*   tags       = (const int*)d_in[2];
  const int*   edges      = (const int*)d_in[3];
  const int*   bridge_ids = (const int*)d_in[4];
  const float* embed_w    = (const float*)d_in[5];
  const float* conv1_w    = (const float*)d_in[6];
  const float* conv1_b    = (const float*)d_in[7];
  const float* conv2_w    = (const float*)d_in[8];
  const float* conv2_b    = (const float*)d_in[9];
  const float* bn1g = (const float*)d_in[10];
  const float* bn1b = (const float*)d_in[11];
  const float* bn1m = (const float*)d_in[12];
  const float* bn1v = (const float*)d_in[13];
  const float* fc1_w = (const float*)d_in[14];
  const float* fc2_w = (const float*)d_in[15];
  const float* bn2g = (const float*)d_in[16];
  const float* bn2b = (const float*)d_in[17];
  const float* bn2m = (const float*)d_in[18];
  const float* bn2v = (const float*)d_in[19];
  const float* gcn1_w = (const float*)d_in[20];
  const float* gcn1_b = (const float*)d_in[21];
  const float* gcn2_w = (const float*)d_in[22];
  const float* gcn2_b = (const float*)d_in[23];

  const int BK = in_sizes[0];              // 512*16
  const int N  = in_sizes[1] / LDESC;      // 30000
  const int NE = in_sizes[3] / 2;          // 480000
  const int NB = in_sizes[4];              // 1000
  const int VE = in_sizes[5];              // V*E

  const int* esrc = edges;
  const int* edst = edges + NE;

  // --- workspace carve-up (all buffers fully re-initialized each call) ---
  char* ws = (char*)d_ws;
  size_t off = 0;
  auto alloc = [&](size_t bytes) -> char* {
    off = (off + 255) & ~(size_t)255;
    char* p = ws + off; off += bytes; return p;
  };
  _Float16* embed_h = (_Float16*)alloc((size_t)VE * 2);
  _Float16* w1h  = (_Float16*)alloc((size_t)F1F2 * 256 * 2);
  _Float16* w2h  = (_Float16*)alloc((size_t)F1F2 * 512 * 2);
  _Float16* fcwh = (_Float16*)alloc((size_t)DCAT * 128 * 2);
  _Float16* g1t  = (_Float16*)alloc((size_t)DMID * DCAT * 2);
  _Float16* g2t  = (_Float16*)alloc((size_t)DOUT * DMID * 2);
  _Float16* hh   = (_Float16*)alloc((size_t)N * DCAT * 2);  // x f16 -> h1 f16
  float* pbuf = (float*)alloc((size_t)N * DCAT * 4);        // hp1 -> hp2
  float* abuf = (float*)alloc((size_t)N * DCAT * 4);        // agg1/h1 -> agg2/h2
  float* dinv = (float*)alloc((size_t)N * 4);

  auto cdiv = [](int a, int b) { return (a + b - 1) / b; };

  // 1) f16 conversions (weights + embed table stay L2-resident)
  cvt_f16_kernel<<<cdiv(VE, 256), 256, 0, stream>>>(embed_w, embed_h, VE);
  cvt_f16_kernel<<<cdiv(F1F2 * 256, 256), 256, 0, stream>>>(conv1_w, w1h, F1F2 * 256);
  cvt_f16_kernel<<<cdiv(F1F2 * 512, 256), 256, 0, stream>>>(conv2_w, w2h, F1F2 * 512);
  cvt_f16_kernel<<<cdiv(F1F2 * 128, 256), 256, 0, stream>>>(fc1_w, fcwh, F1F2 * 128);
  cvt_f16_kernel<<<cdiv(F1F2 * 128, 256), 256, 0, stream>>>(fc2_w, fcwh + (size_t)F1F2 * 128, F1F2 * 128);
  cvt_f16_t_kernel<<<cdiv(DCAT * DMID, 256), 256, 0, stream>>>(gcn1_w, g1t, DCAT, DMID);
  cvt_f16_t_kernel<<<cdiv(DMID * DOUT, 256), 256, 0, stream>>>(gcn2_w, g2t, DMID, DOUT);

  // 2) fused TextCNN + tag branch -> x[N,192] (f16)
  node_features_kernel<<<N, 256, 0, stream>>>(
      feats, tags, embed_h, w1h, w2h, fcwh, conv1_b, conv2_b,
      bn1g, bn1b, bn1m, bn1v, bn2g, bn2b, bn2m, bn2v, hh);

  // 3) degrees / dinv
  deg_init_kernel<<<cdiv(N, 256), 256, 0, stream>>>(dinv, N);
  deg_scatter_kernel<<<cdiv(NE, 256), 256, 0, stream>>>(edst, dinv, NE);
  rsqrt_kernel<<<cdiv(N, 256), 256, 0, stream>>>(dinv, N);

  // 4) GCN layer 1
  {
    int tiles = cdiv(N, 16) * (DMID / 32);
    gcn_gemm_kernel<<<cdiv(tiles, 8), 256, 0, stream>>>(hh, g1t, pbuf, N, DCAT, DMID);
  }
  agg_init_kernel<<<cdiv(N * DMID, 256), 256, 0, stream>>>(abuf, pbuf, dinv, gcn1_b, N, DMID);
  edge_scatter_kernel<<<cdiv(NE, 8), 256, 0, stream>>>(pbuf, abuf, esrc, edst, dinv, NE, DMID);
  relu_cvt_kernel<<<cdiv(N * DMID, 256), 256, 0, stream>>>(abuf, hh, N * DMID);

  // 5) GCN layer 2
  {
    int tiles = cdiv(N, 16) * (DOUT / 32);
    gcn_gemm_kernel<<<cdiv(tiles, 8), 256, 0, stream>>>(hh, g2t, pbuf, N, DMID, DOUT);
  }
  agg_init_kernel<<<cdiv(N * DOUT, 256), 256, 0, stream>>>(abuf, pbuf, dinv, gcn2_b, N, DOUT);
  edge_scatter_kernel<<<cdiv(NE, 8), 256, 0, stream>>>(pbuf, abuf, esrc, edst, dinv, NE, DOUT);

  // 6) L2 normalize + output gathers
  l2norm_kernel<<<cdiv(N, 8), 256, 0, stream>>>(abuf, N);
  gather_out_kernel<<<cdiv((NB + BK) * DOUT, 256), 256, 0, stream>>>(
      abuf, bridge_ids, repo_index, (float*)d_out, NB, BK);
}
